// Mamba2_41068477285015
// MI455X (gfx1250) — matrix operand order repaired and checked by reading, not compiled, not run on hardware
//
#include <hip/hip_runtime.h>
#include <hip/hip_bf16.h>
#include <math.h>

#define D_MODEL   1024
#define D_STATE   64
#define D_CONV    4
#define D_INNER   2048
#define NHEADS    32
#define HEADDIM   64
#define CHUNK     64
#define CONV_DIM  2176      // D_INNER + 2*D_STATE
#define D_IN_PROJ 4256      // 2*D_INNER + 2*D_STATE + NHEADS
#define B_SZ      2
#define T_LEN     4096
#define NCHUNK    (T_LEN / CHUNK)
#define BT        (B_SZ * T_LEN)

typedef __bf16 bf16;
typedef __attribute__((ext_vector_type(16))) __bf16 v16bf;
typedef __attribute__((ext_vector_type(8)))  float  v8f;
typedef __attribute__((ext_vector_type(4)))  unsigned int v4u;
typedef __attribute__((ext_vector_type(8)))  unsigned int v8u;

__device__ __forceinline__ float sigmoidf_(float x) { return 1.f / (1.f + __expf(-x)); }

// ---------------- WMMA fragment helpers (16x16x32 bf16, wave32) ----------------
// A fragment: lane%16 = row M; lanes 0-15 hold K {0..7,16..23}, lanes 16-31 hold
// K {8..15,24..31} (ISA 7.12.2, 16-bit A 16x32). bf16 LDS tile, row length 64.
__device__ __forceinline__ v16bf afrag(const bf16 (*T)[64], int rt, int kk, int lane) {
  int mm = lane & 15;
  int k0 = (lane >> 4) * 8;
  const bf16* r = &T[rt * 16 + mm][kk];
  v16bf a;
  #pragma unroll
  for (int i = 0; i < 8; ++i) { a[i] = r[k0 + i]; a[8 + i] = r[16 + k0 + i]; }
  return a;
}
// A fragment from an fp32 tile with per-row scale (row scale is lane-uniform).
__device__ __forceinline__ v16bf afrag_f32s(const float* base, int ld, int rt, int kk,
                                            int lane, float scale) {
  int mm = lane & 15;
  int k0 = (lane >> 4) * 8;
  const float* r = base + (size_t)(rt * 16 + mm) * ld + kk;
  v16bf a;
  #pragma unroll
  for (int i = 0; i < 8; ++i) {
    a[i]     = (bf16)(r[k0 + i] * scale);
    a[8 + i] = (bf16)(r[16 + k0 + i] * scale);
  }
  return a;
}
// B fragment: lane%16 = column N; lanes 0-15 hold K {0..7,16..23}, lanes 16-31 K {8..15,24..31}.
// T[n][k] = Bmat[k][n] (rows K-contiguous). bf16 LDS tile, row length 64.
__device__ __forceinline__ v16bf bfrag(const bf16 (*T)[64], int nt, int kk, int lane) {
  int nn = lane & 15;
  int kb = (lane >> 4) * 8;
  const bf16* r = &T[nt * 16 + nn][kk];
  v16bf bv;
  #pragma unroll
  for (int i = 0; i < 8; ++i) { bv[i] = r[kb + i]; bv[8 + i] = r[16 + kb + i]; }
  return bv;
}
// B fragment converting from fp32 LDS tile (generic row length).
__device__ __forceinline__ v16bf bfrag_f32(const float* base, int ld, int nt, int kk, int lane) {
  int nn = lane & 15;
  int kb = (lane >> 4) * 8;
  const float* r = base + (size_t)(nt * 16 + nn) * ld + kk;
  v16bf bv;
  #pragma unroll
  for (int i = 0; i < 8; ++i) { bv[i] = (bf16)r[kb + i]; bv[8 + i] = (bf16)r[16 + kb + i]; }
  return bv;
}

// ---------------- fp32 -> bf16 conversion ----------------
__global__ void k_cvt(const float* __restrict__ s, bf16* __restrict__ d, int n) {
  int i = blockIdx.x * 256 + threadIdx.x;
  if (i < n) d[i] = (bf16)s[i];
}

// ---------------- GEMM: C[M,N] = A[M,K] * B[N,K]^T (bf16 in, fp32 out) --------
// 128x64 block tile, 8 waves each computing a 2x2 grid of 16x16 WMMA tiles
// (every fragment reused twice). Double-buffered async staging: issue stage i+1,
// s_wait_asynccnt 6 drains stage i (6 async ops per wave per stage, uniform).
__global__ __launch_bounds__(256) void k_gemm(
    const bf16* __restrict__ A, const bf16* __restrict__ Bm,
    float* __restrict__ Cout, int M, int N, int K)
{
  __shared__ __align__(16) bf16 As[2][128][64];
  __shared__ __align__(16) bf16 Bs[2][64][64];
  const int tid = threadIdx.x, lane = tid & 31, wave = tid >> 5;
  const int wr = wave >> 1, wc = wave & 1;
  const int m = lane & 15, hi = lane >> 4;
  const int rowA0 = blockIdx.y * 128, rowB0 = blockIdx.x * 64;
  // A staging: 2 threads/row, 32 elems (4x16B) each
  const int sar = tid >> 1, sac = (tid & 1) * 32;
  // B staging: 4 threads/row, 16 elems (2x16B) each
  const int sbr = tid >> 2, sbc = (tid & 3) * 16;
  // OOB B rows clamp to a valid row: garbage only feeds cols >= N (never stored),
  // and keeps per-wave async issue count uniform (6 per stage).
  const int rb = rowB0 + sbr;
  const int rbs = (rb < N) ? rb : (N - 1);

  const bf16* gA = A + (size_t)(rowA0 + sar) * K + sac;
  const bf16* gB = Bm + (size_t)rbs * K + sbc;
  unsigned int ldsA[2] = { (unsigned int)(size_t)&As[0][sar][sac],
                           (unsigned int)(size_t)&As[1][sar][sac] };
  unsigned int ldsB[2] = { (unsigned int)(size_t)&Bs[0][sbr][sbc],
                           (unsigned int)(size_t)&Bs[1][sbr][sbc] };
  const int nsteps = K >> 6;

  auto stage = [&](int buf, int kk) {
    const bf16* pa = gA + kk;
    const bf16* pb = gB + kk;
    asm volatile("global_load_async_to_lds_b128 %0, %1, off"           :: "v"(ldsA[buf]), "v"(pa) : "memory");
    asm volatile("global_load_async_to_lds_b128 %0, %1, off offset:16" :: "v"(ldsA[buf]), "v"(pa) : "memory");
    asm volatile("global_load_async_to_lds_b128 %0, %1, off offset:32" :: "v"(ldsA[buf]), "v"(pa) : "memory");
    asm volatile("global_load_async_to_lds_b128 %0, %1, off offset:48" :: "v"(ldsA[buf]), "v"(pa) : "memory");
    asm volatile("global_load_async_to_lds_b128 %0, %1, off"           :: "v"(ldsB[buf]), "v"(pb) : "memory");
    asm volatile("global_load_async_to_lds_b128 %0, %1, off offset:16" :: "v"(ldsB[buf]), "v"(pb) : "memory");
  };

  v8f acc[2][2] = {};
  stage(0, 0);
  for (int step = 0; step < nsteps; ++step) {
    const int cur = step & 1, nxt = cur ^ 1;
    if (step + 1 < nsteps) {
      stage(nxt, (step + 1) * 64);
      asm volatile("s_wait_asynccnt 0x6" ::: "memory");   // stage 'cur' landed
    } else {
      asm volatile("s_wait_asynccnt 0x0" ::: "memory");
    }
    __syncthreads();
    #pragma unroll
    for (int kl = 0; kl < 64; kl += 32) {
      v16bf a0 = afrag(As[cur], wr * 2,     kl, lane);
      v16bf a1 = afrag(As[cur], wr * 2 + 1, kl, lane);
      v16bf b0 = bfrag(Bs[cur], wc * 2,     kl, lane);
      v16bf b1 = bfrag(Bs[cur], wc * 2 + 1, kl, lane);
      acc[0][0] = __builtin_amdgcn_wmma_f32_16x16x32_bf16(false, a0, false, b0, (short)0, acc[0][0], false, false);
      acc[0][1] = __builtin_amdgcn_wmma_f32_16x16x32_bf16(false, a0, false, b1, (short)0, acc[0][1], false, false);
      acc[1][0] = __builtin_amdgcn_wmma_f32_16x16x32_bf16(false, a1, false, b0, (short)0, acc[1][0], false, false);
      acc[1][1] = __builtin_amdgcn_wmma_f32_16x16x32_bf16(false, a1, false, b1, (short)0, acc[1][1], false, false);
    }
    __syncthreads();   // all reads of 'cur' done before it is restaged
  }
  #pragma unroll
  for (int i = 0; i < 2; ++i) {
    #pragma unroll
    for (int j = 0; j < 2; ++j) {
      #pragma unroll
      for (int r = 0; r < 8; ++r) {
        int row = rowA0 + (wr * 2 + i) * 16 + r + hi * 8;
        int col = rowB0 + (wc * 2 + j) * 16 + m;
        if (col < N) Cout[(size_t)row * N + col] = acc[i][j][r];
      }
    }
  }
  (void)M;
}

// ---------------- dt = softplus(raw + bias) ----------------
__global__ void k_dt_softplus(const float* __restrict__ zx,
                              const float* __restrict__ dt_bias,
                              float* __restrict__ dt)
{
  int i = blockIdx.x * 256 + threadIdx.x;
  if (i >= BT * NHEADS) return;
  int h = i % NHEADS;
  size_t bt = (size_t)(i / NHEADS);
  float v = zx[bt * D_IN_PROJ + D_INNER + CONV_DIM + h] + dt_bias[h];
  dt[i] = (v > 20.f) ? v : log1pf(__expf(v));
}

// ---------------- depthwise causal conv(4) + SiLU ----------------
__global__ void k_conv_silu(const float* __restrict__ zx,
                            const float* __restrict__ w,
                            const float* __restrict__ bias,
                            float* __restrict__ xbc)
{
  int i = blockIdx.x * 256 + threadIdx.x;
  if (i >= BT * CONV_DIM) return;
  int cch = i % CONV_DIM;
  int bt = i / CONV_DIM;
  int t = bt % T_LEN, bb = bt / T_LEN;
  float acc = bias[cch];
  #pragma unroll
  for (int j = 0; j < D_CONV; ++j) {
    int tt = t - (D_CONV - 1) + j;
    if (tt >= 0)
      acc += w[cch * D_CONV + j] * zx[(size_t)(bb * T_LEN + tt) * D_IN_PROJ + D_INNER + cch];
  }
  xbc[i] = acc * sigmoidf_(acc);
}

// ---------------- SSD: sequential chunk scan, one block per (b,h) -------------
// Double-buffered TDM staging: wave0 issues chunk c+1's two tensor_load_to_lds
// while all waves compute chunk c; s_wait_tensorcnt 2 drains chunk c's tiles.
__global__ __launch_bounds__(256) void k_ssd(
    const float* __restrict__ xbc, const float* __restrict__ dt,
    const float* __restrict__ A_log, float* __restrict__ yssd)
{
  __shared__ __align__(16) bf16 SL[64][64];        // (l,s) masked decayed scores
  __shared__ __align__(16) bf16 BdT[64][64];       // (n,s): exp(cumA63-cumA[s])*B[s,n]
  __shared__ __align__(16) bf16 XdT[64][64];       // (p,s): x[s,p]*dt[s]
  __shared__ float state[64][64];                  // (p,n) fp32 carry
  __shared__ __align__(16) float rawBC[2][64][128];// TDM: [s][0:64)=B, [64:128)=C
  __shared__ __align__(16) float rawX[2][64][64];  // TDM: x chunk (s,p)
  __shared__ float cumA[64];
  __shared__ float dtc[64];

  const int h = blockIdx.x % NHEADS;
  const int b = blockIdx.x / NHEADS;
  const int tid = threadIdx.x;
  const int lane = tid & 31, wave = tid >> 5;
  const int ti = wave >> 1;
  const int tj0 = (wave & 1) * 2, tj1 = tj0 + 1;
  const int m = lane & 15, hi = lane >> 4;

  const float A_h = -__expf(A_log[h]);
  const unsigned int ldsBC[2] = { (unsigned int)(size_t)&rawBC[0][0][0],
                                  (unsigned int)(size_t)&rawBC[1][0][0] };
  const unsigned int ldsX[2]  = { (unsigned int)(size_t)&rawX[0][0][0],
                                  (unsigned int)(size_t)&rawX[1][0][0] };

  // Issue this chunk's two TDM tile loads (wave0 only; EXEC is ignored by TDM).
  auto tdm_issue = [&](int buf, int chunk) {
    const int t0 = chunk * CHUNK;
    unsigned long long gBC = (unsigned long long)(size_t)
        (xbc + (size_t)(b * T_LEN + t0) * CONV_DIM + D_INNER);
    unsigned long long gX = (unsigned long long)(size_t)
        (xbc + (size_t)(b * T_LEN + t0) * CONV_DIM + h * HEADDIM);
    v4u d0; v8u d1;
    // D# group0: count=1 | lds_addr | global_addr | type=2 (ISA 8.3)
    d0[0] = 1u;
    d0[1] = ldsBC[buf];
    d0[2] = (unsigned int)gBC;
    d0[3] = (unsigned int)(gBC >> 32) | (2u << 30);
    // D# group1 (ISA 8.4): data_size=4B; tensor 128x64; tile 128x64; stride CONV_DIM
    d1[0] = (2u << 16);
    d1[1] = (128u << 16);           // tensor_dim0[15:0] (group bit 48)
    d1[2] = (64u << 16);            // tensor_dim0 hi | tensor_dim1 lo
    d1[3] = (128u << 16);           // tensor_dim1 hi | tile_dim0
    d1[4] = 64u;                    // tile_dim1 | tile_dim2=0
    d1[5] = (unsigned int)CONV_DIM; // tensor_dim0_stride lo32
    d1[6] = 0u;
    d1[7] = 0u;
    asm volatile("tensor_load_to_lds %0, %1" :: "s"(d0), "s"(d1) : "memory");
    // x tile: 64x64 fp32, same row stride
    d0[1] = ldsX[buf];
    d0[2] = (unsigned int)gX;
    d0[3] = (unsigned int)(gX >> 32) | (2u << 30);
    d1[1] = (64u << 16);            // tensor_dim0 = 64
    d1[3] = (64u << 16);            // tile_dim0 = 64
    asm volatile("tensor_load_to_lds %0, %1" :: "s"(d0), "s"(d1) : "memory");
  };

  for (int i = tid; i < 64 * 64; i += 256) state[i >> 6][i & 63] = 0.f;
  if (wave == 0) tdm_issue(0, 0);
  __syncthreads();

  for (int c = 0; c < NCHUNK; ++c) {
    const int t0 = c * CHUNK;
    const int cur = c & 1, nxt = cur ^ 1;
    // prefetch next chunk's raw tiles (buffer nxt was released by chunk c-1)
    if (wave == 0 && c + 1 < NCHUNK) tdm_issue(nxt, c + 1);
    if (tid < 64) dtc[tid] = dt[(size_t)(b * T_LEN + t0 + tid) * NHEADS + h];
    __syncthreads();
    if (tid == 0) {
      float run = 0.f;
      for (int l = 0; l < 64; ++l) { run += A_h * dtc[l]; cumA[l] = run; }
    }
    __syncthreads();
    if (wave == 0) {
      if (c + 1 < NCHUNK) __builtin_amdgcn_s_wait_tensorcnt(2); // chunk c landed
      else                __builtin_amdgcn_s_wait_tensorcnt(0);
    }
    __syncthreads();   // raw tiles + cumA now visible to all waves

    const float (*rBC)[128] = rawBC[cur];
    const float (*rX)[64]   = rawX[cur];

    {   // stage transposed+scaled bf16 tiles (16 contiguous cols per thread)
      int r = tid >> 2;
      int cb = (tid & 3) * 16;
      float dcs = __expf(cumA[63] - cumA[r]);
      float dtv = dtc[r];
      #pragma unroll
      for (int i = 0; i < 16; ++i) {
        int cc = cb + i;
        BdT[cc][r] = (bf16)(rBC[r][cc] * dcs);
        XdT[cc][r] = (bf16)(rX[r][cc] * dtv);
      }
    }
    __syncthreads();

    // G1: S = C @ B^T (contraction over n) straight from the raw TDM tile
    v8f s0 = {}, s1 = {};
    #pragma unroll
    for (int kk = 0; kk < 64; kk += 32) {
      v16bf a = afrag_f32s(&rBC[0][64], 128, ti, kk, lane, 1.f);
      s0 = __builtin_amdgcn_wmma_f32_16x16x32_bf16(false, a, false, bfrag_f32(&rBC[0][0], 128, tj0, kk, lane), (short)0, s0, false, false);
      s1 = __builtin_amdgcn_wmma_f32_16x16x32_bf16(false, a, false, bfrag_f32(&rBC[0][0], 128, tj1, kk, lane), (short)0, s1, false, false);
    }
    // causal mask + decay -> SL
    #pragma unroll
    for (int r = 0; r < 8; ++r) {
      int l = ti * 16 + r + hi * 8;
      float el = cumA[l];
      int sc0 = tj0 * 16 + m;
      int sc1 = sc0 + 16;
      float v0 = (sc0 <= l) ? s0[r] * __expf(el - cumA[sc0]) : 0.f;
      float v1 = (sc1 <= l) ? s1[r] * __expf(el - cumA[sc1]) : 0.f;
      SL[l][sc0] = (bf16)v0;
      SL[l][sc1] = (bf16)v1;
    }
    __syncthreads();

    // G2 + G3: Y = SL @ Xd + (exp(cumA[l]).*C) @ state^T
    v8f y0 = {}, y1 = {};
    #pragma unroll
    for (int kk = 0; kk < 64; kk += 32) {
      v16bf a = afrag(SL, ti, kk, lane);
      y0 = __builtin_amdgcn_wmma_f32_16x16x32_bf16(false, a, false, bfrag(XdT, tj0, kk, lane), (short)0, y0, false, false);
      y1 = __builtin_amdgcn_wmma_f32_16x16x32_bf16(false, a, false, bfrag(XdT, tj1, kk, lane), (short)0, y1, false, false);
    }
    {
      float ecl = __expf(cumA[ti * 16 + m]);  // lane-uniform row scale for A frag
      #pragma unroll
      for (int kk = 0; kk < 64; kk += 32) {
        v16bf a = afrag_f32s(&rBC[0][64], 128, ti, kk, lane, ecl);
        y0 = __builtin_amdgcn_wmma_f32_16x16x32_bf16(false, a, false, bfrag_f32(&state[0][0], 64, tj0, kk, lane), (short)0, y0, false, false);
        y1 = __builtin_amdgcn_wmma_f32_16x16x32_bf16(false, a, false, bfrag_f32(&state[0][0], 64, tj1, kk, lane), (short)0, y1, false, false);
      }
    }
    #pragma unroll
    for (int r = 0; r < 8; ++r) {
      int l = ti * 16 + r + hi * 8;
      size_t base = ((size_t)(b * T_LEN + t0 + l) * NHEADS + h) * HEADDIM;
      yssd[base + tj0 * 16 + m] = y0[r];
      yssd[base + tj1 * 16 + m] = y1[r];
    }

    // G4: contrib[p,n] = sum_s Xd[s,p] * (decay[s]*B[s,n])
    v8f u0 = {}, u1 = {};
    #pragma unroll
    for (int kk = 0; kk < 64; kk += 32) {
      v16bf a = afrag(XdT, ti, kk, lane);
      u0 = __builtin_amdgcn_wmma_f32_16x16x32_bf16(false, a, false, bfrag(BdT, tj0, kk, lane), (short)0, u0, false, false);
      u1 = __builtin_amdgcn_wmma_f32_16x16x32_bf16(false, a, false, bfrag(BdT, tj1, kk, lane), (short)0, u1, false, false);
    }
    __syncthreads();   // G3's state reads complete before update
    {
      float dl = __expf(cumA[63]);
      #pragma unroll
      for (int r = 0; r < 8; ++r) {
        int p = ti * 16 + r + hi * 8;
        int n0 = tj0 * 16 + m;
        state[p][n0]      = state[p][n0] * dl + u0[r];
        state[p][n0 + 16] = state[p][n0 + 16] * dl + u1[r];
      }
    }
    __syncthreads();   // chunk c's raw tiles free for chunk c+2's TDM
  }
}

// ---------------- y = (Y + x*D) * silu(z); RMSNorm; -> bf16 -------------------
__global__ __launch_bounds__(256) void k_gate_norm(
    const float* __restrict__ yssd, const float* __restrict__ xbc,
    const float* __restrict__ zx, const float* __restrict__ Dv,
    const float* __restrict__ norm_w, bf16* __restrict__ yn)
{
  const int bt = blockIdx.x;
  const float* yrow = yssd + (size_t)bt * D_INNER;
  const float* xrow = xbc + (size_t)bt * CONV_DIM;
  const float* zrow = zx + (size_t)bt * D_IN_PROJ;
  float vals[D_INNER / 256];
  float local = 0.f;
  #pragma unroll
  for (int j = 0; j < D_INNER / 256; ++j) {
    int d = threadIdx.x + j * 256;
    int h = d >> 6;   // HEADDIM = 64
    float y = yrow[d] + xrow[d] * Dv[h];
    float z = zrow[d];
    y *= z * sigmoidf_(z);
    vals[j] = y;
    local += y * y;
  }
  #pragma unroll
  for (int off2 = 16; off2 > 0; off2 >>= 1) local += __shfl_xor(local, off2, 32);
  __shared__ float red[8];
  __shared__ float sscale;
  if ((threadIdx.x & 31) == 0) red[threadIdx.x >> 5] = local;
  __syncthreads();
  if (threadIdx.x == 0) {
    float s = 0.f;
    for (int i = 0; i < 8; ++i) s += red[i];
    sscale = rsqrtf(s / (float)D_INNER + 1e-5f);
  }
  __syncthreads();
  float sc = sscale;
  #pragma unroll
  for (int j = 0; j < D_INNER / 256; ++j) {
    int d = threadIdx.x + j * 256;
    yn[(size_t)bt * D_INNER + d] = (bf16)(vals[j] * sc * norm_w[d]);
  }
}

// ---------------- launch ----------------
extern "C" void kernel_launch(void* const* d_in, const int* in_sizes, int n_in,
                              void* d_out, int out_size, void* d_ws, size_t ws_size,
                              hipStream_t stream)
{
  const float* u       = (const float*)d_in[0];
  const float* W_in    = (const float*)d_in[1];
  const float* conv_w  = (const float*)d_in[2];
  const float* conv_b  = (const float*)d_in[3];
  const float* dt_bias = (const float*)d_in[4];
  const float* A_log   = (const float*)d_in[5];
  const float* Dv      = (const float*)d_in[6];
  const float* norm_w  = (const float*)d_in[7];
  const float* W_out   = (const float*)d_in[8];
  float* out = (float*)d_out;

  char* ws = (char*)d_ws;
  size_t off = 0;
  auto wsalloc = [&](size_t bytes) -> char* {
    char* p = ws + off;
    off += (bytes + 255) & ~(size_t)255;
    return p;
  };
  bf16*  ubf    = (bf16*) wsalloc((size_t)BT * D_MODEL * 2);
  bf16*  winbf  = (bf16*) wsalloc((size_t)D_IN_PROJ * D_MODEL * 2);
  bf16*  woutbf = (bf16*) wsalloc((size_t)D_MODEL * D_INNER * 2);
  float* zx     = (float*)wsalloc((size_t)BT * D_IN_PROJ * 4);
  float* xbc    = (float*)wsalloc((size_t)BT * CONV_DIM * 4);
  float* dtp    = (float*)wsalloc((size_t)BT * NHEADS * 4);
  float* yssd   = (float*)wsalloc((size_t)BT * D_INNER * 4);
  bf16*  ynbf   = (bf16*) wsalloc((size_t)BT * D_INNER * 2);

  k_cvt<<<(BT * D_MODEL + 255) / 256, 256, 0, stream>>>(u, ubf, BT * D_MODEL);
  k_cvt<<<(D_IN_PROJ * D_MODEL + 255) / 256, 256, 0, stream>>>(W_in, winbf, D_IN_PROJ * D_MODEL);
  k_cvt<<<(D_MODEL * D_INNER + 255) / 256, 256, 0, stream>>>(W_out, woutbf, D_MODEL * D_INNER);

  k_gemm<<<dim3((D_IN_PROJ + 63) / 64, BT / 128), 256, 0, stream>>>(ubf, winbf, zx, BT, D_IN_PROJ, D_MODEL);
  k_dt_softplus<<<(BT * NHEADS + 255) / 256, 256, 0, stream>>>(zx, dt_bias, dtp);
  k_conv_silu<<<(BT * CONV_DIM + 255) / 256, 256, 0, stream>>>(zx, conv_w, conv_b, xbc);
  k_ssd<<<B_SZ * NHEADS, 256, 0, stream>>>(xbc, dtp, A_log, yssd);
  k_gate_norm<<<BT, 256, 0, stream>>>(yssd, xbc, zx, Dv, norm_w, ynbf);
  k_gemm<<<dim3(D_MODEL / 64, BT / 128), 256, 0, stream>>>(ynbf, woutbf, out, BT, D_MODEL, D_INNER);

  (void)in_sizes; (void)n_in; (void)out_size; (void)ws_size;
}
